// DPB_Attention_12214886990360
// MI455X (gfx1250) — matrix operand order repaired and checked by reading, not compiled, not run on hardware
//
#include <hip/hip_runtime.h>

// ---------------------------------------------------------------------------
// DPB attention for MI455X (gfx1250), wave32 + WMMA f16->f32 + TDM staging.
// B=256, C=256, H=W=16, N=256, heads=8, d=32, L=(2H-1)(2W-1)=961, pd=16.
// ---------------------------------------------------------------------------

typedef __attribute__((ext_vector_type(16))) _Float16 v16h;
typedef __attribute__((ext_vector_type(8)))  _Float16 v8h;
typedef __attribute__((ext_vector_type(8)))  float    v8f;
typedef unsigned int uint32x4 __attribute__((ext_vector_type(4)));
typedef int          int32x4  __attribute__((ext_vector_type(4)));
typedef int          int32x8  __attribute__((ext_vector_type(8)));

#define NHEADS 8
#define DIM    256
#define NN     256
#define DHEAD  32
#define LREL   961
#define PD     16

#if defined(__has_builtin)
#if __has_builtin(__builtin_amdgcn_tensor_load_to_lds)
#define HAVE_TDM 1
#endif
#if __has_builtin(__builtin_amdgcn_s_wait_tensorcnt)
#define HAVE_TCNT 1
#endif
#endif

__device__ __forceinline__ v8f wmma_f16(v16h a, v16h b, v8f c) {
  // D = A(16x32 f16) * B(32x16 f16) + C(16x16 f32)
  return __builtin_amdgcn_wmma_f32_16x16x32_f16(false, a, false, b, (short)0, c,
                                                false, false);
}

// WMMA 16-bit A-fragment element mapping (ISA 7.12.2):
//   lane row = lane&15 ; element i -> K = ((i&8)<<1) + ((lane>>4)<<3) + (i&7)
// This file's B (and A) fragments use contiguous K instead:
//   lane col = lane&15 ; element j -> K = ((lane>>4)<<4) + j   (contiguous!)

// Build an A fragment from two contiguous 8-half chunks in memory
// (row-major source, K innermost): chunks at k0+kh and k0+kh+8, kh=(lane>>4)*16.
__device__ __forceinline__ v16h frag_a_rowmajor(const _Float16* rowp, int kh) {
  const v8h c0 = *reinterpret_cast<const v8h*>(rowp + kh);
  const v8h c1 = *reinterpret_cast<const v8h*>(rowp + kh + 8);
  v16h a;
#pragma unroll
  for (int i = 0; i < 8; ++i) { a[i] = c0[i]; a[i + 8] = c1[i]; }
  return a;
}

// ---------------------------------------------------------------------------
// Kernel 1: dynamic-position-bias MLP.  961 rows: (dy,dx) -> 16 -> 16 -> 16 -> 8
// ---------------------------------------------------------------------------
__device__ __forceinline__ void ln_relu16(float* v, const float* g, const float* b) {
  float mu = 0.f;
#pragma unroll
  for (int i = 0; i < PD; ++i) mu += v[i];
  mu *= (1.0f / PD);
  float var = 0.f;
#pragma unroll
  for (int i = 0; i < PD; ++i) { const float d = v[i] - mu; var += d * d; }
  var *= (1.0f / PD);
  const float inv = rsqrtf(var + 1e-5f);
#pragma unroll
  for (int i = 0; i < PD; ++i) {
    const float t = (v[i] - mu) * inv * g[i] + b[i];
    v[i] = t > 0.f ? t : 0.f;
  }
}

__global__ void k_dpb_mlp(const float* __restrict__ pw,  const float* __restrict__ pb,
                          const float* __restrict__ g1,  const float* __restrict__ b1,
                          const float* __restrict__ f1w, const float* __restrict__ f1b,
                          const float* __restrict__ g2,  const float* __restrict__ b2,
                          const float* __restrict__ f2w, const float* __restrict__ f2b,
                          const float* __restrict__ g3,  const float* __restrict__ b3,
                          const float* __restrict__ f3w, const float* __restrict__ f3b,
                          float* __restrict__ p_table) {
  const int r = blockIdx.x * blockDim.x + threadIdx.x;
  if (r >= LREL) return;
  const float dy = (float)(r / 31 - 15);
  const float dx = (float)(r % 31 - 15);
  float v[PD], t[PD];
#pragma unroll
  for (int i = 0; i < PD; ++i) v[i] = dy * pw[2 * i] + dx * pw[2 * i + 1] + pb[i];
  ln_relu16(v, g1, b1);
#pragma unroll
  for (int o = 0; o < PD; ++o) {
    float s = f1b[o];
#pragma unroll
    for (int i = 0; i < PD; ++i) s += v[i] * f1w[o * PD + i];
    t[o] = s;
  }
  ln_relu16(t, g2, b2);
#pragma unroll
  for (int o = 0; o < PD; ++o) {
    float s = f2b[o];
#pragma unroll
    for (int i = 0; i < PD; ++i) s += t[i] * f2w[o * PD + i];
    v[o] = s;
  }
  ln_relu16(v, g3, b3);
#pragma unroll
  for (int o = 0; o < NHEADS; ++o) {
    float s = f3b[o];
#pragma unroll
    for (int i = 0; i < PD; ++i) s += v[i] * f3w[o * PD + i];
    p_table[r * NHEADS + o] = s;
  }
}

// ---------------------------------------------------------------------------
// Kernel 2: f32 weights -> f16.
// ---------------------------------------------------------------------------
__global__ void k_cvt_w(const float* __restrict__ qkv_w, const float* __restrict__ proj_w,
                        _Float16* __restrict__ wq, _Float16* __restrict__ wp) {
  const int i = blockIdx.x * 256 + threadIdx.x;
  if (i < 3 * DIM * DIM) wq[i] = (_Float16)qkv_w[i];
  if (i < DIM * DIM)     wp[i] = (_Float16)proj_w[i];
}

// ---------------------------------------------------------------------------
// Kernel 3: qkv GEMM.  A = x[b]^T staged in LDS *in fragment order* (one b128
// LDS read per operand); B fragments are 32B-contiguous -> direct global b128.
// Writes q (scaled), k, v as f16 in (B, h, N, d).
// ---------------------------------------------------------------------------
__global__ void k_qkv_gemm(const float* __restrict__ x, const _Float16* __restrict__ wq,
                           const float* __restrict__ qkv_b,
                           _Float16* __restrict__ qh, _Float16* __restrict__ kh,
                           _Float16* __restrict__ vh) {
  __shared__ alignas(32) _Float16 As[4 * 32 * 16];  // 4 row-tiles x 32 lanes x 16 halves
  const int b      = blockIdx.z;
  const int o_base = blockIdx.x * 64;
  const int n_base = blockIdx.y * 64;
  const int t = threadIdx.x, lane = t & 31, w = t >> 5;
  const int tm = w & 3;              // row tile (4 x 16)
  const int tc = (w >> 2) * 2;       // base col tile (pairs of 16)
  v8f acc0 = {}, acc1 = {};
  const float* xb = x + (size_t)b * DIM * NN;
  const int col0 = o_base + tc * 16 + (lane & 15);
  const int col1 = col0 + 16;
  const int kseg = (lane >> 4) << 4;

  for (int k0 = 0; k0 < DIM; k0 += 32) {
    if (k0 + 32 < DIM)
      __builtin_prefetch(xb + (size_t)(k0 + 32) * NN + n_base + (t & 63), 0, 1);
    // A[nn][cc] = x[b, k0+cc, n_base+nn]  -> fragment-order slot (coalesced read)
    for (int e = t; e < 2048; e += 256) {
      const int nn = e & 63, cc = e >> 6;
      const int l = (nn & 15) | (((cc >> 4) & 1) << 4);
      const int i = cc & 15;
      As[((((nn >> 4) << 5) | l) << 4) + i] =
          (_Float16)xb[(size_t)(k0 + cc) * NN + n_base + nn];
    }
    __syncthreads();
    const v16h a  = *reinterpret_cast<const v16h*>(&As[(tm * 32 + lane) * 16]);
    const v16h b0 = *reinterpret_cast<const v16h*>(wq + (size_t)col0 * DIM + k0 + kseg);
    const v16h b1 = *reinterpret_cast<const v16h*>(wq + (size_t)col1 * DIM + k0 + kseg);
    acc0 = wmma_f16(a, b0, acc0);
    acc1 = wmma_f16(a, b1, acc1);
    __syncthreads();
  }

  const float scale = 0.17677669529663687f;  // 32^-0.5
#pragma unroll
  for (int half = 0; half < 2; ++half) {
    const v8f acc = half ? acc1 : acc0;
    const int o = o_base + (tc + half) * 16 + (lane & 15);
    const int which = o >> 8;            // 0=q 1=k 2=v
    const int head  = (o >> 5) & 7;
    const int dd    = o & 31;
    _Float16* dst = which == 0 ? qh : (which == 1 ? kh : vh);
    const float fb = qkv_b[o];
#pragma unroll
    for (int r = 0; r < 8; ++r) {
      const int m = n_base + tm * 16 + ((lane >> 4) << 3) + r;
      float val = acc[r] + fb;
      if (which == 0) val *= scale;
      dst[(((size_t)b * NHEADS + head) * NN + m) * DHEAD + dd] = (_Float16)val;
    }
  }
}

// ---------------------------------------------------------------------------
// Kernel 4: attention.  Dynamic LDS (64 KB): S 32K | V 16K | P 16K.
// V staged by the Tensor Data Mover (one TENSOR_LOAD_TO_LDS, TENSORcnt fence);
// P written by softmax in WMMA fragment order -> phase-3 A loads are b128.
// ---------------------------------------------------------------------------
#define VSH_OFF 32768
#define PC_OFF  49152

__global__ void k_attn(const _Float16* __restrict__ qh, const _Float16* __restrict__ kg_,
                       const _Float16* __restrict__ vg_, const float* __restrict__ p_table,
                       _Float16* __restrict__ ao) {
  extern __shared__ char smem[];
  float*    S   = reinterpret_cast<float*>(smem);               // 32 KB scores f32
  _Float16* vsh = reinterpret_cast<_Float16*>(smem + VSH_OFF);  // 16 KB V[256][32]
  _Float16* Pc  = reinterpret_cast<_Float16*>(smem + PC_OFF);   // 16 KB P (frag order)
  const int qb0 = blockIdx.x * 32;
  const int h   = blockIdx.y;
  const int b   = blockIdx.z;
  const int t = threadIdx.x, lane = t & 31, w = t >> 5;
  const size_t bh = (size_t)b * NHEADS + h;
  const _Float16* qg = qh  + bh * NN * DHEAD;
  const _Float16* kg = kg_ + bh * NN * DHEAD;
  const _Float16* vg = vg_ + bh * NN * DHEAD;

  // ---- Stage V[256][32] f16 (16 KB linear tile) via TDM ----
#if defined(HAVE_TDM)
  if (w == 0) {
    const unsigned long long ga = (unsigned long long)(uintptr_t)vg;
    uint32x4 g0;
    g0[0] = 1u;                                   // count=1, user descriptor
    g0[1] = (unsigned)VSH_OFF;                    // lds_addr (bytes)
    g0[2] = (unsigned)(ga & 0xffffffffu);         // global_addr[31:0]
    g0[3] = (unsigned)((ga >> 32) & 0x01ffffffu)  // global_addr[56:32]
            | 0x80000000u;                        // type=2 ("image")
    int32x8 g1;
    g1[0] = 0x00010000;          // data_size=1 (2 bytes/elem)
    g1[1] = (int)(8192u << 16);  // tensor_dim0[15:0]=8192 @ bits 63:48
    g1[2] = (int)(1u << 16);     // tensor_dim0 hi=0 | tensor_dim1=1
    g1[3] = (int)(8192u << 16);  // tile_dim0=8192 @ bits 127:112
    g1[4] = 1;                   // tile_dim1=1, tile_dim2=0
    g1[5] = 8192;                // tensor_dim0_stride
    g1[6] = 0;
    g1[7] = 0;
    const int32x4 z4 = {0, 0, 0, 0};
#if __clang_major__ >= 23
    const int32x8 z8 = {0, 0, 0, 0, 0, 0, 0, 0};
    __builtin_amdgcn_tensor_load_to_lds(g0, g1, z4, z4, z8, 0);
#else
    __builtin_amdgcn_tensor_load_to_lds(g0, g1, z4, z4, 0);
#endif
  }
#else
  {
    const uint4* src = (const uint4*)vg;
    uint4* dst = (uint4*)vsh;
    for (int e = t; e < 1024; e += 256) dst[e] = src[e];
  }
#endif

  // ---- Phase 1: S = q * k^T + rel_bias ----
  const int tm = w & 1;
  const v16h a = frag_a_rowmajor(qg + (size_t)(qb0 + tm * 16 + (lane & 15)) * DHEAD,
                                 (lane >> 4) << 4);
#pragma unroll
  for (int cti = 0; cti < 4; ++cti) {
    const int ct   = (w >> 1) + cti * 4;        // key tile 0..15
    const int jcol = ct * 16 + (lane & 15);     // key index (B col)
    // B fragment = k^T tile: one 32B aligned b128x2 load per lane
    const v16h bf = *reinterpret_cast<const v16h*>(
        kg + (size_t)jcol * DHEAD + ((lane >> 4) << 4));
    v8f acc = {};
    acc = wmma_f16(a, bf, acc);
#pragma unroll
    for (int r = 0; r < 8; ++r) {
      const int ml = tm * 16 + ((lane >> 4) << 3) + r;
      const int i  = qb0 + ml;                  // global query token
      const int j  = jcol;                      // global key token
      const int rel = ((i >> 4) - (j >> 4) + 15) * 31 + ((i & 15) - (j & 15) + 15);
      S[ml * 256 + j] = acc[r] + p_table[rel * NHEADS + h];
    }
  }
  __syncthreads();

  // ---- Phase 2: row softmax (one wave per 4 rows); write P in fragment order:
  //      tile (ptm=r>>4, kj=c>>5), lane = (r&15)|((c>>4&1)<<4), i = c&15
#pragma unroll 1
  for (int rr = 0; rr < 4; ++rr) {
    const int row = w * 4 + rr;
    const float* Sr = &S[row * 256];
    float e[8];
    float mx = -3.4e38f;
#pragma unroll
    for (int u = 0; u < 8; ++u) mx = fmaxf(mx, Sr[lane + 32 * u]);
#pragma unroll
    for (int off = 16; off > 0; off >>= 1) mx = fmaxf(mx, __shfl_xor(mx, off, 32));
    float sum = 0.f;
#pragma unroll
    for (int u = 0; u < 8; ++u) { e[u] = __expf(Sr[lane + 32 * u] - mx); sum += e[u]; }
#pragma unroll
    for (int off = 16; off > 0; off >>= 1) sum += __shfl_xor(sum, off, 32);
    const float inv = 1.0f / sum;
#pragma unroll
    for (int u = 0; u < 8; ++u) {
      const int c = lane + 32 * u;
      const int l = (row & 15) | (((c >> 4) & 1) << 4);
      const int i = c & 15;
      Pc[(((((row >> 4) << 3) | (c >> 5)) << 5 | l) << 4) + i] = (_Float16)(e[u] * inv);
    }
  }
#if defined(HAVE_TDM) && defined(HAVE_TCNT)
  if (w == 0) __builtin_amdgcn_s_wait_tensorcnt(0);  // V tile landed in LDS
#endif
  __syncthreads();

  // ---- Phase 3: O = P * V  (waves 0..3, one 16x16 tile each, full K) ----
  if (w < 4) {
    const int ptm = w & 1, ptn = w >> 1;
    const int col = ptn * 16 + (lane & 15);
    const int c0  = (lane >> 4) << 4;
    v8f acc = {};
#pragma unroll 1
    for (int kj = 0; kj < 8; ++kj) {
      // A fragment: one contiguous 32B LDS read (fragment-ordered P)
      const v16h pa = *reinterpret_cast<const v16h*>(
          &Pc[((ptm * 8 + kj) * 32 + lane) * 16]);
      v16h vb;
#pragma unroll
      for (int j2 = 0; j2 < 16; ++j2) vb[j2] = vsh[(kj * 32 + c0 + j2) * 32 + col];
      acc = wmma_f16(pa, vb, acc);
    }
#pragma unroll
    for (int r = 0; r < 8; ++r) {
      const int n  = qb0 + ptm * 16 + ((lane >> 4) << 3) + r;
      const int dd = ptn * 16 + (lane & 15);
      ao[((size_t)b * NN + n) * DIM + h * DHEAD + dd] = (_Float16)acc[r];
    }
  }
}

// ---------------------------------------------------------------------------
// Kernel 5: output projection GEMM, fully LDS-free: A and B fragments are
// contiguous in HBM -> direct b128 loads. Transposed f32 store to (B,C,H,W).
// ---------------------------------------------------------------------------
__global__ void k_proj_gemm(const _Float16* __restrict__ ao, const _Float16* __restrict__ wp,
                            const float* __restrict__ proj_b, float* __restrict__ out) {
  const int b      = blockIdx.z;
  const int o_base = blockIdx.x * 64;
  const int n_base = blockIdx.y * 64;
  const int t = threadIdx.x, lane = t & 31, w = t >> 5;
  const int tm = w & 3, tc = (w >> 2) * 2;
  v8f acc0 = {}, acc1 = {};
  const _Float16* ab = ao + (size_t)b * NN * DIM;
  const int row  = n_base + tm * 16 + (lane & 15);
  const int khf  = (lane >> 4) << 4;
  const int col0 = o_base + tc * 16 + (lane & 15);
  const int col1 = col0 + 16;
  const int kseg = (lane >> 4) << 4;

#pragma unroll 2
  for (int k0 = 0; k0 < DIM; k0 += 32) {
    const v16h a  = frag_a_rowmajor(ab + (size_t)row * DIM + k0, khf);
    const v16h b0 = *reinterpret_cast<const v16h*>(wp + (size_t)col0 * DIM + k0 + kseg);
    const v16h b1 = *reinterpret_cast<const v16h*>(wp + (size_t)col1 * DIM + k0 + kseg);
    acc0 = wmma_f16(a, b0, acc0);
    acc1 = wmma_f16(a, b1, acc1);
  }

#pragma unroll
  for (int half = 0; half < 2; ++half) {
    const v8f acc = half ? acc1 : acc0;
    const int o = o_base + (tc + half) * 16 + (lane & 15);
    const float fb = proj_b[o];
#pragma unroll
    for (int r = 0; r < 8; ++r) {
      const int m = n_base + tm * 16 + ((lane >> 4) << 3) + r;   // token -> (h,w)
      out[(size_t)b * DIM * NN + (size_t)o * NN + m] = acc[r] + fb;
    }
  }
}

// ---------------------------------------------------------------------------
// Launch
// ---------------------------------------------------------------------------
extern "C" void kernel_launch(void* const* d_in, const int* in_sizes, int n_in,
                              void* d_out, int out_size, void* d_ws, size_t ws_size,
                              hipStream_t stream) {
  (void)in_sizes; (void)n_in; (void)out_size; (void)ws_size;
  const float* x      = (const float*)d_in[0];
  const float* qkv_w  = (const float*)d_in[1];
  const float* qkv_b  = (const float*)d_in[2];
  const float* proj_w = (const float*)d_in[3];
  const float* proj_b = (const float*)d_in[4];
  const float* pos_w  = (const float*)d_in[5];
  const float* pos_b  = (const float*)d_in[6];
  const float* g1 = (const float*)d_in[7];   const float* b1 = (const float*)d_in[8];
  const float* f1w = (const float*)d_in[9];  const float* f1b = (const float*)d_in[10];
  const float* g2 = (const float*)d_in[11];  const float* b2 = (const float*)d_in[12];
  const float* f2w = (const float*)d_in[13]; const float* f2b = (const float*)d_in[14];
  const float* g3 = (const float*)d_in[15];  const float* b3 = (const float*)d_in[16];
  const float* f3w = (const float*)d_in[17]; const float* f3b = (const float*)d_in[18];

  // Workspace layout (all offsets 256B aligned):
  char* ws = (char*)d_ws;
  const size_t SZ_QKV = (size_t)256 * NHEADS * NN * DHEAD * sizeof(_Float16); // 33.5 MB
  float*    p_table = (float*)(ws);                                  // 961*8 f32
  _Float16* wq_h    = (_Float16*)(ws + 32768);                       // 768*256 f16
  _Float16* wp_h    = (_Float16*)(ws + 32768 + 393216);              // 256*256 f16
  _Float16* qh      = (_Float16*)(ws + 557056);
  _Float16* kh      = (_Float16*)(ws + 557056 + SZ_QKV);
  _Float16* vh      = (_Float16*)(ws + 557056 + 2 * SZ_QKV);
  _Float16* aoh     = (_Float16*)(ws + 557056 + 3 * SZ_QKV);

  k_dpb_mlp<<<dim3(31), dim3(32), 0, stream>>>(pos_w, pos_b, g1, b1, f1w, f1b,
                                               g2, b2, f2w, f2b, g3, b3, f3w, f3b,
                                               p_table);
  k_cvt_w<<<dim3(768), dim3(256), 0, stream>>>(qkv_w, proj_w, wq_h, wp_h);
  k_qkv_gemm<<<dim3(12, 4, 256), dim3(256), 0, stream>>>(x, wq_h, qkv_b, qh, kh, vh);
  k_attn<<<dim3(8, 8, 256), dim3(256), 65536, stream>>>(qh, kh, vh, p_table, aoh);
  k_proj_gemm<<<dim3(4, 4, 256), dim3(256), 0, stream>>>(aoh, wp_h, proj_b,
                                                         (float*)d_out);
}
